// VQEmbedding_44478681317657
// MI455X (gfx1250) — matrix operand order repaired and checked by reading, not compiled, not run on hardware
//
#include <hip/hip_runtime.h>

typedef __attribute__((ext_vector_type(16))) __bf16 v16bf;
typedef __attribute__((ext_vector_type(8)))  float  v8f;

#define DDIM   512
#define KCODES 1024
#define NTOK   (8 * 2048)
#define TOKW   32                 // tokens per wave (2 slabs of 16)
#define WPB    2                  // waves per block
#define NWAVES (NTOK / TOKW)      // 512 loss partials

union ABReg { uint4 q[2]; v16bf v; };
union BFU   { __bf16 h; unsigned short u; };

static __device__ __forceinline__ unsigned short f2bf(float f) {
    BFU b; b.h = (__bf16)f; return b.u;
}
static __device__ __forceinline__ float bf2f(unsigned short u) {
    BFU b; b.u = u; return (float)b.h;
}
static __device__ __forceinline__ unsigned pack2(float a, float b) {
    return (unsigned)f2bf(a) | ((unsigned)f2bf(b) << 16);
}
// 8 fp32 -> 8 bf16 hi + 8 bf16 lo (one uint4 each)
static __device__ __forceinline__ void cvt8(const float4 a, const float4 b,
                                            uint4& hi, uint4& lo) {
    hi.x = pack2(a.x, a.y); hi.y = pack2(a.z, a.w);
    hi.z = pack2(b.x, b.y); hi.w = pack2(b.z, b.w);
    lo.x = pack2(a.x - bf2f((unsigned short)(hi.x & 0xffff)),
                 a.y - bf2f((unsigned short)(hi.x >> 16)));
    lo.y = pack2(a.z - bf2f((unsigned short)(hi.y & 0xffff)),
                 a.w - bf2f((unsigned short)(hi.y >> 16)));
    lo.z = pack2(b.x - bf2f((unsigned short)(hi.z & 0xffff)),
                 b.y - bf2f((unsigned short)(hi.z >> 16)));
    lo.w = pack2(b.z - bf2f((unsigned short)(hi.w & 0xffff)),
                 b.w - bf2f((unsigned short)(hi.w >> 16)));
}

// ---------------------------------------------------------------------------
// Kernel 1: split codebook W (f32) into bf16 hi/lo planes + exact ||w_k||^2.
// ---------------------------------------------------------------------------
__global__ __launch_bounds__(128) void vq_prep(
    const float* __restrict__ W,
    unsigned short* __restrict__ Whi,
    unsigned short* __restrict__ Wlo,
    float* __restrict__ w2)
{
    const int k    = blockIdx.x;
    const int t    = threadIdx.x;
    const int lane = t & 31, wave = t >> 5;

    const float4 v = *(const float4*)(W + ((size_t)k << 9) + t * 4);

    uint2 ph, pl;
    ph.x = pack2(v.x, v.y); ph.y = pack2(v.z, v.w);
    pl.x = pack2(v.x - bf2f((unsigned short)(ph.x & 0xffff)),
                 v.y - bf2f((unsigned short)(ph.x >> 16)));
    pl.y = pack2(v.z - bf2f((unsigned short)(ph.y & 0xffff)),
                 v.w - bf2f((unsigned short)(ph.y >> 16)));
    *(uint2*)(Whi + ((size_t)k << 9) + t * 4) = ph;
    *(uint2*)(Wlo + ((size_t)k << 9) + t * 4) = pl;

    float s = v.x * v.x + v.y * v.y + v.z * v.z + v.w * v.w;
    #pragma unroll
    for (int m = 1; m < 32; m <<= 1) s += __shfl_xor(s, m, 32);

    __shared__ float acc4[4];
    if (lane == 0) acc4[wave] = s;
    __syncthreads();
    if (t == 0) w2[k] = acc4[0] + acc4[1] + acc4[2] + acc4[3];
}

// ---------------------------------------------------------------------------
// Kernel 2: each wave owns 32 tokens (two 16-row WMMA slabs, A held entirely
// in VGPRs as bf16 hi/lo) vs all 1024 codes streamed from L2. bf16x3 WMMA
// argmin, exact fp32 gather + per-wave loss partial. No LDS, no barriers.
// ---------------------------------------------------------------------------
__global__ __launch_bounds__(64) void vq_main(
    const float* __restrict__ x,
    const float* __restrict__ W,
    const unsigned short* __restrict__ Whi,
    const unsigned short* __restrict__ Wlo,
    const float* __restrict__ w2,
    float* __restrict__ outq,
    float* __restrict__ partials)
{
    const int lane = threadIdx.x & 31;
    const int wave = threadIdx.x >> 5;
    const int rl   = lane & 15;
    const int kab  = (lane < 16) ? 0 : 16;   // A K-base (contiguous k = 16h + s, matches B)
    const int kbb  = (lane < 16) ? 0 : 16;   // B K-base (contiguous k = 16h + s, matches A)
    const size_t wrow0 = ((size_t)blockIdx.x * WPB + wave) * TOKW;

    // ---- preamble: load x rows straight into the WMMA A layout, fp32->bf16
    // hi/lo split in VALU. 2 slabs x 16 depth-chunks x 16 halves per lane.
    ABReg ahi[2][16], alo[2][16];
    #pragma unroll
    for (int s = 0; s < 2; ++s) {
        const float* xr = x + (wrow0 + (size_t)s * 16 + rl) * DDIM;
        #pragma unroll
        for (int c = 0; c < 16; ++c) {
            const int d0 = c * 32 + kab;
            const float4 a0 = *(const float4*)(xr + d0);
            const float4 a1 = *(const float4*)(xr + d0 + 4);
            const float4 b0 = *(const float4*)(xr + d0 + 8);
            const float4 b1 = *(const float4*)(xr + d0 + 12);
            cvt8(a0, a1, ahi[s][c].q[0], alo[s][c].q[0]);
            cvt8(b0, b1, ahi[s][c].q[1], alo[s][c].q[1]);
        }
    }

    float bestd[2][8];
    int   besti[2][8];
    #pragma unroll
    for (int s = 0; s < 2; ++s)
        #pragma unroll
        for (int j = 0; j < 8; ++j) { bestd[s][j] = 3.402823e38f; besti[s][j] = 0; }

    const v8f vzero = {0.f, 0.f, 0.f, 0.f, 0.f, 0.f, 0.f, 0.f};

    for (int cg = 0; cg < KCODES / 32; ++cg) {       // 32 groups of 32 codes
        const int cbase = cg * 32;

        // prefetch next group's B rows into the cache hierarchy
        if (cg + 1 < KCODES / 32) {
            const size_t ncode = (size_t)(cbase + 32 + lane);
            __builtin_prefetch(Whi + (ncode << 9), 0, 0);
            __builtin_prefetch(Wlo + (ncode << 9), 0, 0);
        }

        v8f acc[2][2];
        #pragma unroll
        for (int s = 0; s < 2; ++s) { acc[s][0] = vzero; acc[s][1] = vzero; }

        #pragma unroll
        for (int c = 0; c < 16; ++c) {               // depth chunks of 32
            const int d0 = c << 5;
            #pragma unroll
            for (int jj = 0; jj < 2; ++jj) {         // 2 column tiles of 16 codes
                const int    code = cbase + jj * 16 + rl;
                const size_t boff = ((size_t)code << 9) + d0 + kbb;
                ABReg bhi, blo;
                const uint4* bph = (const uint4*)(Whi + boff);
                const uint4* bpl = (const uint4*)(Wlo + boff);
                bhi.q[0] = bph[0]; bhi.q[1] = bph[1];
                blo.q[0] = bpl[0]; blo.q[1] = bpl[1];
                // bf16x3 emulated-fp32: hi*hi + hi*lo + lo*hi, both A slabs
                // share this B fetch (341 B of L2 traffic per WMMA).
                #pragma unroll
                for (int s = 0; s < 2; ++s) {
                    acc[s][jj] = __builtin_amdgcn_wmma_f32_16x16x32_bf16(
                        false, ahi[s][c].v, false, bhi.v, (short)0, acc[s][jj], false, false);
                    acc[s][jj] = __builtin_amdgcn_wmma_f32_16x16x32_bf16(
                        false, ahi[s][c].v, false, blo.v, (short)0, acc[s][jj], false, false);
                    acc[s][jj] = __builtin_amdgcn_wmma_f32_16x16x32_bf16(
                        false, alo[s][c].v, false, bhi.v, (short)0, acc[s][jj], false, false);
                }
            }
        }

        // argmin epilogue: d2 = ||w||^2 - 2*x.w  (x^2 term irrelevant to argmin)
        #pragma unroll
        for (int jj = 0; jj < 2; ++jj) {
            const int   code = cbase + jj * 16 + rl;
            const float wk   = w2[code];
            #pragma unroll
            for (int s = 0; s < 2; ++s)
                #pragma unroll
                for (int j = 0; j < 8; ++j) {
                    const float dist = __builtin_fmaf(-2.0f, acc[s][jj][j], wk);
                    if (dist < bestd[s][j]) { bestd[s][j] = dist; besti[s][j] = code; }
                }
        }
    }

    // reduce (min,idx) across the 16 lanes holding each token; ties -> lower idx
    #pragma unroll
    for (int s = 0; s < 2; ++s)
        #pragma unroll
        for (int j = 0; j < 8; ++j)
            #pragma unroll
            for (int m = 1; m < 16; m <<= 1) {
                const float od = __shfl_xor(bestd[s][j], m, 32);
                const int   oi = __shfl_xor(besti[s][j], m, 32);
                if (od < bestd[s][j] || (od == bestd[s][j] && oi < besti[s][j])) {
                    bestd[s][j] = od; besti[s][j] = oi;
                }
            }

    // gather exact fp32 codeword rows + accumulate (q-x)^2
    float lsum = 0.f;
    #pragma unroll
    for (int m = 0; m < TOKW; ++m) {
        const int s  = m >> 4;
        const int mm = m & 15;
        const int idx = __shfl(besti[s][mm & 7], (mm < 8) ? 0 : 16, 32);
        const size_t tok = wrow0 + m;
        const float* wr = W + ((size_t)idx << 9);
        const float* xr = x + tok * DDIM;
        float*       qr = outq + tok * DDIM;
        const int db = lane * 16;
        #pragma unroll
        for (int v4 = 0; v4 < 4; ++v4) {
            const float4 wv = *(const float4*)(wr + db + v4 * 4);
            const float4 xv = *(const float4*)(xr + db + v4 * 4);
            *(float4*)(qr + db + v4 * 4) = wv;
            const float e0 = wv.x - xv.x, e1 = wv.y - xv.y;
            const float e2 = wv.z - xv.z, e3 = wv.w - xv.w;
            lsum += e0 * e0 + e1 * e1 + e2 * e2 + e3 * e3;
        }
    }
    #pragma unroll
    for (int m = 1; m < 32; m <<= 1) lsum += __shfl_xor(lsum, m, 32);
    if (lane == 0) partials[(size_t)blockIdx.x * WPB + wave] = lsum;  // plain store
}

// ---------------------------------------------------------------------------
// Kernel 3: fixed-order reduction of loss partials -> scalar loss.
// ---------------------------------------------------------------------------
__global__ __launch_bounds__(32) void vq_loss(
    const float* __restrict__ partials, float* __restrict__ loss_out)
{
    float s = 0.f;
    for (int i = threadIdx.x; i < NWAVES; i += 32) s += partials[i];
    #pragma unroll
    for (int m = 1; m < 32; m <<= 1) s += __shfl_xor(s, m, 32);
    if (threadIdx.x == 0)
        *loss_out = s * (1.25f / (float)((size_t)NTOK * DDIM));  // q_loss + 0.25*e_loss
}

extern "C" void kernel_launch(void* const* d_in, const int* in_sizes, int n_in,
                              void* d_out, int out_size, void* d_ws, size_t ws_size,
                              hipStream_t stream) {
    (void)in_sizes; (void)n_in; (void)out_size; (void)ws_size;
    const float* x = (const float*)d_in[0];
    const float* W = (const float*)d_in[1];
    float* outq     = (float*)d_out;
    float* loss_out = outq + (size_t)NTOK * DDIM;

    char* ws = (char*)d_ws;                                   // layout (~2.06 MB):
    unsigned short* Whi = (unsigned short*)ws;                //  1 MB bf16 hi
    unsigned short* Wlo = (unsigned short*)(ws + (size_t)KCODES * DDIM * 2); // 1 MB bf16 lo
    float* w2       = (float*)(ws + (size_t)KCODES * DDIM * 4);              // 4 KB
    float* partials = w2 + KCODES;                                           // 2 KB

    hipLaunchKernelGGL(vq_prep, dim3(KCODES), dim3(128), 0, stream, W, Whi, Wlo, w2);
    hipLaunchKernelGGL(vq_main, dim3(NTOK / (TOKW * WPB)), dim3(WPB * 32), 0, stream,
                       x, W, Whi, Wlo, w2, outq, partials);
    hipLaunchKernelGGL(vq_loss, dim3(1), dim3(32), 0, stream, partials, loss_out);
}
